// UMABlock_46531675685338
// MI455X (gfx1250) — matrix operand order repaired and checked by reading, not compiled, not run on hardware
//
#include <hip/hip_runtime.h>
#include <math.h>

typedef __bf16  v16bf __attribute__((ext_vector_type(16)));
typedef float   v8f   __attribute__((ext_vector_type(8)));
typedef unsigned short u16;

// ---------------- small helpers ----------------
__device__ __forceinline__ u16 f2bf(float f) {
    unsigned int u = __float_as_uint(f);
    u = (u + 0x7FFFu + ((u >> 16) & 1u)) >> 16;
    return (u16)u;
}
__device__ __forceinline__ float siluf(float x) { return x / (1.f + __expf(-x)); }
__device__ __forceinline__ float sigmf(float x) { return 1.f / (1.f + __expf(-x)); }

__device__ __forceinline__ float blockSum128(float v, float* sm) {
    int t = threadIdx.x;
    sm[t] = v; __syncthreads();
    for (int s = 64; s > 0; s >>= 1) { if (t < s) sm[t] += sm[t + s]; __syncthreads(); }
    float r = sm[0]; __syncthreads();
    return r;
}

// B2 basis (5 x 3 x 3)
__constant__ float B2c[45] = {
    0.f, 0.70710678f, 0.f,  0.70710678f, 0.f, 0.f,  0.f, 0.f, 0.f,
    0.f, 0.f, 0.f,  0.f, 0.f, 0.70710678f,  0.f, 0.70710678f, 0.f,
    -0.40824829f, 0.f, 0.f,  0.f, -0.40824829f, 0.f,  0.f, 0.f, 0.81649658f,
    0.f, 0.f, 0.70710678f,  0.f, 0.f, 0.f,  0.70710678f, 0.f, 0.f,
    0.70710678f, 0.f, 0.f,  0.f, -0.70710678f, 0.f,  0.f, 0.f, 0.f
};

// ---------------- WMMA GEMM: C = A(MxK,bf16) * B (given as Bt: NxK bf16) ----------------
// Requires M % 16 == 0, N % 64 == 0, K % 32 == 0 (true for every GEMM in this net).
// Each wave computes a 16x64 output strip: one A fragment reused across 4 B tiles.
union ABu { v16bf v; uint4 u4[2]; };

__global__ void k_gemm_bt(const u16* __restrict__ A, const u16* __restrict__ Bt,
                          const float* __restrict__ bias,
                          float* __restrict__ Cf, u16* __restrict__ Cb,
                          int M, int N, int K, int act)
{
    int lane = threadIdx.x & 31;
    int wid  = blockIdx.x * 8 + (threadIdx.x >> 5);
    int NT4  = N >> 6;
    int MT   = M >> 4;
    if (wid >= MT * NT4) return;
    int mt = wid / NT4, nt = wid % NT4;
    int m0 = mt << 4, n0 = nt << 6;
    int half = lane >> 4, l15 = lane & 15;

    const u16* ap = A  + (size_t)(m0 + l15) * K + half * 16;  // A frag: 16 consecutive K of row m
    const u16* bp = Bt + (size_t)(n0 + l15) * K + half * 16;  // B frag: 16 consecutive K of column n
    const size_t bs = (size_t)16 * K;

    v8f acc0 = {0.f,0.f,0.f,0.f,0.f,0.f,0.f,0.f};
    v8f acc1 = {0.f,0.f,0.f,0.f,0.f,0.f,0.f,0.f};
    v8f acc2 = {0.f,0.f,0.f,0.f,0.f,0.f,0.f,0.f};
    v8f acc3 = {0.f,0.f,0.f,0.f,0.f,0.f,0.f,0.f};

    for (int k = 0; k < K; k += 32) {
        ABu a, b0, b1, b2, b3;
        a.u4[0]  = *(const uint4*)(ap + k);
        a.u4[1]  = *(const uint4*)(ap + k + 8);
        b0.u4[0] = *(const uint4*)(bp + k);
        b0.u4[1] = *(const uint4*)(bp + k + 8);
        b1.u4[0] = *(const uint4*)(bp + bs + k);
        b1.u4[1] = *(const uint4*)(bp + bs + k + 8);
        b2.u4[0] = *(const uint4*)(bp + 2 * bs + k);
        b2.u4[1] = *(const uint4*)(bp + 2 * bs + k + 8);
        b3.u4[0] = *(const uint4*)(bp + 3 * bs + k);
        b3.u4[1] = *(const uint4*)(bp + 3 * bs + k + 8);
        acc0 = __builtin_amdgcn_wmma_f32_16x16x32_bf16(false, a.v, false, b0.v, (short)0, acc0, false, false);
        acc1 = __builtin_amdgcn_wmma_f32_16x16x32_bf16(false, a.v, false, b1.v, (short)0, acc1, false, false);
        acc2 = __builtin_amdgcn_wmma_f32_16x16x32_bf16(false, a.v, false, b2.v, (short)0, acc2, false, false);
        acc3 = __builtin_amdgcn_wmma_f32_16x16x32_bf16(false, a.v, false, b3.v, (short)0, acc3, false, false);
    }

    int mbase = m0 + half * 8;
    v8f accs[4] = { acc0, acc1, acc2, acc3 };
    for (int t = 0; t < 4; t++) {
        int nn = n0 + t * 16 + l15;
        float bv = bias ? bias[nn] : 0.f;
        for (int r = 0; r < 8; r++) {
            int m = mbase + r;
            float v = accs[t][r] + bv;
            if (act) v = siluf(v);
            if (Cf) Cf[(size_t)m * N + nn] = v;
            if (Cb) Cb[(size_t)m * N + nn] = f2bf(v);
        }
    }
}

// ---------------- weight prep ----------------
// dst (N x K) bf16 = transpose of src (K x N) f32
__global__ void k_transcvt(u16* dst, const float* src, int K, int N) {
    int idx = blockIdx.x * 256 + threadIdx.x;
    if (idx >= N * K) return;
    int n = idx / K, k = idx % K;
    dst[idx] = f2bf(src[(size_t)k * N + n]);
}
// combined complex weight: Wc = [[wr, wi], [-wi, wr]] (2K x 2N); stored transposed (2N x 2K)
__global__ void k_combine(u16* dst, const float* wr, const float* wi, int K, int N) {
    int idx = blockIdx.x * 256 + threadIdx.x;
    int tot = 4 * K * N;
    if (idx >= tot) return;
    int j = idx / (2 * K), kk = idx % (2 * K);
    float v;
    if (j < N)  v = (kk < K) ? wr[(size_t)kk * N + j]        : -wi[(size_t)(kk - K) * N + j];
    else { int jj = j - N; v = (kk < K) ? wi[(size_t)kk * N + jj] :  wr[(size_t)(kk - K) * N + jj]; }
    dst[idx] = f2bf(v);
}

// ---------------- YG (64x9) and YGI = pinv (9x64) on device ----------------
__global__ void k_prep_ygi(float* YGo, float* YGIo) {
    __shared__ float Yg[64][9];
    __shared__ float Mi[9][9];
    int p = threadIdx.x; // 64 threads
    {
        int bi = p >> 3, ai = p & 7;
        float bet = 3.14159265358979f * (bi + 0.5f) / 8.f;
        float alp = 6.28318530717959f * ai / 8.f;
        float yv = cosf(bet), sb = sinf(bet);
        float xv = sb * sinf(alp), zv = sb * cosf(alp);
        float r[3] = { xv, yv, zv };
        Yg[p][0] = 1.f;
        const float s3 = 1.7320508075688772f, s5 = 2.2360679774997896f;
        Yg[p][1] = yv * s3; Yg[p][2] = zv * s3; Yg[p][3] = xv * s3;
        for (int m = 0; m < 5; m++) {
            float s = 0.f;
            for (int i = 0; i < 3; i++) for (int j = 0; j < 3; j++) s += B2c[m*9+i*3+j] * r[i] * r[j];
            Yg[p][4 + m] = s * s5;
        }
    }
    __syncthreads();
    if (p == 0) {
        float Ma[9][18];
        for (int i = 0; i < 9; i++)
            for (int j = 0; j < 9; j++) {
                float s = 0.f;
                for (int q = 0; q < 64; q++) s += Yg[q][i] * Yg[q][j];
                Ma[i][j] = s;
            }
        for (int i = 0; i < 9; i++) for (int j = 0; j < 9; j++) Ma[i][9 + j] = (i == j) ? 1.f : 0.f;
        for (int col = 0; col < 9; col++) {
            int piv = col; float best = fabsf(Ma[col][col]);
            for (int rr = col + 1; rr < 9; rr++) { float v = fabsf(Ma[rr][col]); if (v > best) { best = v; piv = rr; } }
            if (piv != col) for (int j = 0; j < 18; j++) { float t = Ma[col][j]; Ma[col][j] = Ma[piv][j]; Ma[piv][j] = t; }
            float inv = 1.f / Ma[col][col];
            for (int j = 0; j < 18; j++) Ma[col][j] *= inv;
            for (int rr = 0; rr < 9; rr++) if (rr != col) {
                float f = Ma[rr][col];
                for (int j = 0; j < 18; j++) Ma[rr][j] -= f * Ma[col][j];
            }
        }
        for (int i = 0; i < 9; i++) for (int j = 0; j < 9; j++) Mi[i][j] = Ma[i][9 + j];
    }
    __syncthreads();
    for (int l = 0; l < 9; l++) {
        float s = 0.f;
        for (int j = 0; j < 9; j++) s += Mi[l][j] * Yg[p][j];
        YGIo[l * 64 + p] = s;
    }
    for (int l = 0; l < 9; l++) YGo[p * 9 + l] = Yg[p][l];
}

// ---------------- geometry: D (9x9), envelope, distance ----------------
__device__ __forceinline__ void mul3(const float a[3][3], const float b[3][3], float o[3][3]) {
    for (int i = 0; i < 3; i++) for (int j = 0; j < 3; j++) {
        float s = 0.f;
        for (int k = 0; k < 3; k++) s += a[i][k] * b[k][j];
        o[i][j] = s;
    }
}

__global__ void k_geom(const float* __restrict__ ev, const float* __restrict__ rg,
                       float* __restrict__ Dbuf, float* __restrict__ envb,
                       float* __restrict__ db, int E)
{
    int e = blockIdx.x * blockDim.x + threadIdx.x;
    if (e >= E) return;
    float vx = ev[e*3+0], vy = ev[e*3+1], vz = ev[e*3+2];
    float d = sqrtf(vx*vx + vy*vy + vz*vz + 1e-12f);
    float nx = vx / d, ny = vy / d, nz = vz / d;
    float beta  = acosf(fminf(1.f, fmaxf(-1.f, ny)));
    float alpha = atan2f(nx, nz);
    float gamma = rg[e] * 6.28318530717959f;
    float ca = cosf(alpha), sa = sinf(alpha);
    float cb = cosf(beta),  sb = sinf(beta);
    float cg = cosf(gamma), sg = sinf(gamma);
    float Ra[3][3] = {{ca,0,sa},{0,1,0},{-sa,0,ca}};
    float Rb[3][3] = {{1,0,0},{0,cb,-sb},{0,sb,cb}};
    float Rg[3][3] = {{cg,0,sg},{0,1,0},{-sg,0,cg}};
    float T[3][3], R[3][3];
    mul3(Rb, Rg, T); mul3(Ra, T, R);

    float Dl[81];
    for (int i = 0; i < 81; i++) Dl[i] = 0.f;
    Dl[0] = 1.f;
    const int p3[3] = {1, 2, 0};
    for (int i = 0; i < 3; i++) for (int j = 0; j < 3; j++)
        Dl[(1+i)*9 + (1+j)] = R[p3[i]][p3[j]];
    for (int n = 0; n < 5; n++) {
        float U[3][3], T2[3][3];
        for (int a2 = 0; a2 < 3; a2++) for (int j = 0; j < 3; j++) {
            float s = 0.f;
            for (int b2 = 0; b2 < 3; b2++) s += B2c[n*9 + a2*3 + b2] * R[j][b2];
            U[a2][j] = s;
        }
        for (int i2 = 0; i2 < 3; i2++) for (int j = 0; j < 3; j++) {
            float s = 0.f;
            for (int a2 = 0; a2 < 3; a2++) s += R[i2][a2] * U[a2][j];
            T2[i2][j] = s;
        }
        for (int m = 0; m < 5; m++) {
            float s = 0.f;
            for (int i2 = 0; i2 < 3; i2++) for (int j = 0; j < 3; j++) s += B2c[m*9 + i2*3 + j] * T2[i2][j];
            Dl[(4+m)*9 + (4+n)] = s;
        }
    }
    float* dp = Dbuf + (size_t)e * 81;
    for (int i = 0; i < 81; i++) dp[i] = Dl[i];

    float x = d * 0.2f; // d / CUTOFF
    float x5 = x*x*x*x*x;
    float env = (x < 1.f) ? (1.f - 21.f*x5 + 35.f*x5*x - 15.f*x5*x*x) : 0.f;
    envb[e] = env;
    db[e]   = d;
}

// ---------------- edge embeds: [rbf(512) | send_emb | recv_emb] bf16 ----------------
__global__ void k_embeds(const float* __restrict__ db, const int* __restrict__ senders,
                         const int* __restrict__ receivers, const int* __restrict__ species,
                         const float* __restrict__ send_emb, const float* __restrict__ recv_emb,
                         u16* __restrict__ EE, int E)
{
    int e = blockIdx.x; if (e >= E) return;
    int t = threadIdx.x;
    float d = db[e];
    const float istd = 511.f / 10.f;
    u16* row = EE + (size_t)e * 768;
    for (int j = t; j < 512; j += 128) {
        float z = (d - 5.f * j / 511.f) * istd;
        row[j] = f2bf(__expf(-0.5f * z * z));
    }
    int sp = species[senders[e]];
    int rp = species[receivers[e]];
    row[512 + t] = f2bf(send_emb[sp * 128 + t]);
    row[640 + t] = f2bf(recv_emb[rp * 128 + t]);
}

// ---------------- init x ----------------
__global__ void k_init_x(float* __restrict__ x, const int* __restrict__ species,
                         const float* __restrict__ sphere_emb, const float* __restrict__ csd,
                         const int* __restrict__ n_node, int NB, int Nn)
{
    int n = blockIdx.x; if (n >= Nn) return;
    int c = threadIdx.x;
    int b = 0, cum = 0;
    for (int i = 0; i < NB; i++) { int nn = n_node[i]; if (n >= cum && n < cum + nn) b = i; cum += nn; }
    float v = sphere_emb[(size_t)species[n] * 128 + c] + csd[(size_t)b * 128 + c];
    float* xp = x + (size_t)n * 1152;
    xp[c] = v;
    for (int l = 1; l < 9; l++) xp[l * 128 + c] = 0.f;
}

// ---------------- initial message scatter ----------------
__global__ void k_msg0(const float* __restrict__ m0f, const float* __restrict__ Dbuf,
                       const float* __restrict__ envb, const int* __restrict__ recv,
                       float* __restrict__ x, int E)
{
    int e = blockIdx.x; if (e >= E) return;
    int c = threadIdx.x;
    __shared__ float Ds[81];
    if (c < 81) Ds[c] = Dbuf[(size_t)e * 81 + c];
    __syncthreads();
    const float* mp = m0f + (size_t)e * 384;
    float m0 = mp[c], m1 = mp[128 + c], m2 = mp[256 + c];
    float sc = envb[e] * 0.2f;
    float* xp = x + (size_t)recv[e] * 1152;
    for (int l = 0; l < 9; l++) {
        float v = (Ds[0*9+l]*m0 + Ds[2*9+l]*m1 + Ds[6*9+l]*m2) * sc;
        atomicAdd(xp + l * 128 + c, v);
    }
}

// ---------------- spherical layer norm ----------------
__global__ void k_lnsh(const float* __restrict__ x, float* __restrict__ out,
                       const float* __restrict__ g0, const float* __restrict__ b0,
                       const float* __restrict__ w1, const float* __restrict__ w2, int Nn)
{
    __shared__ float sm[128];
    int n = blockIdx.x; if (n >= Nn) return;
    int c = threadIdx.x;
    const float* xp = x + (size_t)n * 1152;
    float x0 = xp[c];
    float mu = blockSum128(x0, sm) * (1.f / 128.f);
    float dv = x0 - mu;
    float var = blockSum128(dv * dv, sm) * (1.f / 128.f);
    float o0 = dv * rsqrtf(var + 1e-5f) * g0[c] + b0[c];
    float x1 = xp[128+c], x2 = xp[256+c], x3 = xp[384+c];
    float n1 = blockSum128(x1*x1 + x2*x2 + x3*x3, sm) * (1.f / 128.f);
    float s1 = rsqrtf(n1 + 1e-5f) * w1[c];
    float x4 = xp[512+c], x5 = xp[640+c], x6 = xp[768+c], x7 = xp[896+c], x8 = xp[1024+c];
    float n2 = blockSum128(x4*x4 + x5*x5 + x6*x6 + x7*x7 + x8*x8, sm) * (1.f / 128.f);
    float s2 = rsqrtf(n2 + 1e-5f) * w2[c];
    float* op = out + (size_t)n * 1152;
    op[c]        = o0;
    op[128 + c]  = x1 * s1; op[256 + c]  = x2 * s1; op[384 + c]  = x3 * s1;
    op[512 + c]  = x4 * s2; op[640 + c]  = x5 * s2; op[768 + c]  = x6 * s2;
    op[896 + c]  = x7 * s2; op[1024 + c] = x8 * s2;
}

// ---------------- radial second stage: rs = hid @ rad_w2 + rad_b2 ----------------
__global__ void k_rad2(const float* __restrict__ hid, const float* __restrict__ w2,
                       const float* __restrict__ b2, float* __restrict__ rs, int ecnt)
{
    int i = blockIdx.x * blockDim.x + threadIdx.x;
    if (i >= ecnt) return;
    const float* h = hid + (size_t)i * 128;
    for (int l = 0; l < 9; l++) {
        float s = b2[l];
        for (int k = 0; k < 128; k++) s += h[k] * w2[k * 9 + l];
        rs[i * 9 + l] = s;
    }
}

// ---------------- build rotated, rs-scaled SO2 A-operands ----------------
__global__ void k_edgewise(const float* __restrict__ xln, const float* __restrict__ Dbuf,
                           const float* __restrict__ rs, const int* __restrict__ senders,
                           const int* __restrict__ receivers, int e0,
                           u16* __restrict__ A0, u16* __restrict__ A1, u16* __restrict__ A2, int ecnt)
{
    int i = blockIdx.x; if (i >= ecnt) return;
    int e = e0 + i;
    int c = threadIdx.x;
    __shared__ float Ds[81];
    __shared__ float rss[9];
    if (c < 81) Ds[c] = Dbuf[(size_t)e * 81 + c];
    if (c >= 96 && c < 105) rss[c - 96] = rs[i * 9 + (c - 96)];
    __syncthreads();
    const float* xs = xln + (size_t)senders[e]   * 1152 + c;
    const float* xr = xln + (size_t)receivers[e] * 1152 + c;
    float a[9], b[9];
    for (int j = 0; j < 9; j++) { a[j] = xs[j * 128]; b[j] = xr[j * 128]; }
    float ya[9], yb[9];
    for (int l = 0; l < 9; l++) {
        float sa = 0.f, sb = 0.f;
        for (int j = 0; j < 9; j++) { sa += Ds[l*9+j]*a[j]; sb += Ds[l*9+j]*b[j]; }
        ya[l] = sa * rss[l]; yb[l] = sb * rss[l];
    }
    size_t r0 = (size_t)i * 768;   // rows [0, 2, 6], each 256 (xs|xd)
    A0[r0 +   0 + c] = f2bf(ya[0]); A0[r0 + 128 + c] = f2bf(yb[0]);
    A0[r0 + 256 + c] = f2bf(ya[2]); A0[r0 + 384 + c] = f2bf(yb[2]);
    A0[r0 + 512 + c] = f2bf(ya[6]); A0[r0 + 640 + c] = f2bf(yb[6]);
    size_t r1 = (size_t)i * 1024;  // rows [3, 7, 1, 5]
    A1[r1 +   0 + c] = f2bf(ya[3]); A1[r1 + 128 + c] = f2bf(yb[3]);
    A1[r1 + 256 + c] = f2bf(ya[7]); A1[r1 + 384 + c] = f2bf(yb[7]);
    A1[r1 + 512 + c] = f2bf(ya[1]); A1[r1 + 640 + c] = f2bf(yb[1]);
    A1[r1 + 768 + c] = f2bf(ya[5]); A1[r1 + 896 + c] = f2bf(yb[5]);
    size_t r2 = (size_t)i * 512;   // rows [8, 4]
    A2[r2 +   0 + c] = f2bf(ya[8]); A2[r2 + 128 + c] = f2bf(yb[8]);
    A2[r2 + 256 + c] = f2bf(ya[4]); A2[r2 + 384 + c] = f2bf(yb[4]);
}

// ---------------- gate nonlinearity between conv1 and conv2 ----------------
__global__ void k_nonlin(const float* __restrict__ y0f, const float* __restrict__ p1o,
                         const float* __restrict__ p2o,
                         u16* __restrict__ hA0, u16* __restrict__ hA1, u16* __restrict__ hA2, int ecnt)
{
    int i = blockIdx.x; if (i >= ecnt) return;
    int c = threadIdx.x;
    const float* y0 = y0f + (size_t)i * 640;
    const float* p1 = p1o + (size_t)i * 512;
    const float* p2 = p2o + (size_t)i * 256;
    float g1 = sigmf(y0[384 + c]);
    float g2 = sigmf(y0[512 + c]);
    float h0 = siluf(y0[c]);
    float h2 = y0[128 + c] * g1;
    float h6 = y0[256 + c] * g2;
    float h3 = p1[c]       * g1;
    float h7 = p1[128 + c] * g2;
    float h1 = p1[256 + c] * g1;
    float h5 = p1[384 + c] * g2;
    float h8 = p2[c]       * g2;
    float h4 = p2[128 + c] * g2;
    size_t r0 = (size_t)i * 384;
    hA0[r0 + c] = f2bf(h0); hA0[r0 + 128 + c] = f2bf(h2); hA0[r0 + 256 + c] = f2bf(h6);
    size_t r1 = (size_t)i * 512;
    hA1[r1 + c] = f2bf(h3); hA1[r1 + 128 + c] = f2bf(h7);
    hA1[r1 + 256 + c] = f2bf(h1); hA1[r1 + 384 + c] = f2bf(h5);
    size_t r2 = (size_t)i * 256;
    hA2[r2 + c] = f2bf(h8); hA2[r2 + 128 + c] = f2bf(h4);
}

// ---------------- rotate back (Dinv = D^T), envelope, scatter ----------------
__global__ void k_msg_scatter(const float* __restrict__ c2y0, const float* __restrict__ c2p1,
                              const float* __restrict__ c2p2, const float* __restrict__ Dbuf,
                              const float* __restrict__ envb, const int* __restrict__ recv,
                              float* __restrict__ x, int e0, int ecnt)
{
    int i = blockIdx.x; if (i >= ecnt) return;
    int e = e0 + i;
    int c = threadIdx.x;
    __shared__ float Ds[81];
    if (c < 81) Ds[c] = Dbuf[(size_t)e * 81 + c];
    __syncthreads();
    const float* y0 = c2y0 + (size_t)i * 384;
    const float* p1 = c2p1 + (size_t)i * 512;
    const float* p2 = c2p2 + (size_t)i * 256;
    float y[9];
    y[0] = y0[c]; y[2] = y0[128 + c]; y[6] = y0[256 + c];
    y[3] = p1[c]; y[7] = p1[128 + c]; y[1] = p1[256 + c]; y[5] = p1[384 + c];
    y[8] = p2[c]; y[4] = p2[128 + c];
    float enve = envb[e];
    float* xp = x + (size_t)recv[e] * 1152;
    for (int l = 0; l < 9; l++) {
        float s = 0.f;
        for (int j = 0; j < 9; j++) s += Ds[j * 9 + l] * y[j];
        atomicAdd(xp + l * 128 + c, s * enve);
    }
}

// ---------------- atomwise: project onto sphere grid and back ----------------
__global__ void k_yg(const float* __restrict__ xln, int n0, const float* __restrict__ YG,
                     u16* __restrict__ g, int ncnt)
{
    __shared__ float Ys[576];
    int i = blockIdx.x; if (i >= ncnt) return;
    int c = threadIdx.x;
    for (int t = c; t < 576; t += 128) Ys[t] = YG[t];
    __syncthreads();
    const float* xp = xln + (size_t)(n0 + i) * 1152 + c;
    float xl[9];
    for (int l = 0; l < 9; l++) xl[l] = xp[l * 128];
    u16* gp = g + (size_t)i * 64 * 128 + c;
    for (int p = 0; p < 64; p++) {
        float s = 0.f;
        for (int l = 0; l < 9; l++) s += Ys[p * 9 + l] * xl[l];
        gp[p * 128] = f2bf(s);
    }
}

__global__ void k_ygi(const float* __restrict__ gout, const float* __restrict__ YGI,
                      float* __restrict__ x, int n0, int ncnt)
{
    __shared__ float Ys[576];
    int i = blockIdx.x; if (i >= ncnt) return;
    int c = threadIdx.x;
    for (int t = c; t < 576; t += 128) Ys[t] = YGI[t];
    __syncthreads();
    const float* gp = gout + (size_t)i * 64 * 128 + c;
    float acc[9] = {0,0,0,0,0,0,0,0,0};
    for (int p = 0; p < 64; p++) {
        float gv = gp[p * 128];
        for (int l = 0; l < 9; l++) acc[l] += Ys[l * 64 + p] * gv;
    }
    float* xp = x + (size_t)(n0 + i) * 1152 + c;
    for (int l = 0; l < 9; l++) xp[l * 128] += acc[l];
}

// =============================================================================
extern "C" void kernel_launch(void* const* d_in, const int* in_sizes, int n_in,
                              void* d_out, int out_size, void* d_ws, size_t ws_size,
                              hipStream_t stream)
{
    const float* ev        = (const float*)d_in[0];
    const int*   species   = (const int*)  d_in[1];
    const float* csd       = (const float*)d_in[2];
    const int*   senders   = (const int*)  d_in[3];
    const int*   receivers = (const int*)  d_in[4];
    const int*   n_node    = (const int*)  d_in[5];
    const float* rot_gamma = (const float*)d_in[6];
    const int E  = in_sizes[3];
    const int Nn = in_sizes[1];
    const int NB = in_sizes[5];

    auto P  = [&](int i) { return (const float*)d_in[7 + i]; };
    auto LP = [&](int l, int i) { return (const float*)d_in[18 + l * 26 + i]; };
    // params: 0 sphere_emb, 1 send_emb, 2 recv_emb, 3 ed_w1, 4 ed_b1, 5 ed_w2, 6 ed_b2,
    //         7 lnf_g0, 8 lnf_b0, 9 lnf_w1, 10 lnf_w2; layers at 18 + l*26:
    // 0..3 ln1, 4 rad_w1, 5 rad_b1, 6 rad_w2, 7 rad_b2, 8 c1_w0, 9 c1_wr1, 10 c1_wi1,
    // 11 c1_wr2, 12 c1_wi2, 13 c2_w0, 14 c2_wr1, 15 c2_wi1, 16 c2_wr2, 17 c2_wi2,
    // 18..21 ln2, 22 a_w1, 23 a_b1, 24 a_w2, 25 a_b2

    // ---- workspace carve-out ----
    uintptr_t base = (uintptr_t)d_ws;
    size_t off = 0;
    auto alloc = [&](size_t bytes) -> void* {
        off = (off + 255) & ~(size_t)255;
        void* p = (void*)(base + off);
        off += bytes;
        return p;
    };
    float* Dbuf = (float*)alloc((size_t)E * 81 * 4);
    float* envb = (float*)alloc((size_t)E * 4);
    float* dbuf = (float*)alloc((size_t)E * 4);
    u16*   EE   = (u16*)  alloc((size_t)E * 768 * 2);
    float* xln  = (float*)alloc((size_t)Nn * 1152 * 4);
    float* YG   = (float*)alloc(576 * 4);
    float* YGI  = (float*)alloc(576 * 4);
    u16* ed_w1T = (u16*)alloc((size_t)128 * 768 * 2);
    u16* ed_w2T = (u16*)alloc((size_t)384 * 128 * 2);
    u16 *radw1T[2], *c1w0T[2], *c1c1T[2], *c1c2T[2], *c2w0T[2], *c2c1T[2], *c2c2T[2], *aw1T[2], *aw2T[2];
    for (int l = 0; l < 2; l++) {
        radw1T[l] = (u16*)alloc((size_t)128 * 768  * 2);
        c1w0T[l]  = (u16*)alloc((size_t)640 * 768  * 2);
        c1c1T[l]  = (u16*)alloc((size_t)512 * 1024 * 2);
        c1c2T[l]  = (u16*)alloc((size_t)256 * 512  * 2);
        c2w0T[l]  = (u16*)alloc((size_t)384 * 384  * 2);
        c2c1T[l]  = (u16*)alloc((size_t)512 * 512  * 2);
        c2c2T[l]  = (u16*)alloc((size_t)256 * 256  * 2);
        aw1T[l]   = (u16*)alloc((size_t)128 * 128  * 2);
        aw2T[l]   = (u16*)alloc((size_t)128 * 128  * 2);
    }
    const size_t arena0 = (off + 255) & ~(size_t)255;

    float* x = (float*)d_out;   // node features (Nn x 9 x 128), residuals via atomic adds

    auto gemm = [&](const u16* A, const u16* Bt, const float* bias,
                    float* Cf, u16* Cb, int M, int N, int K, int act) {
        long tiles = (long)((M + 15) / 16) * (N / 64);
        int blocks = (int)((tiles + 7) / 8);
        k_gemm_bt<<<blocks, 256, 0, stream>>>(A, Bt, bias, Cf, Cb, M, N, K, act);
    };
    auto tc = [&](u16* dst, const float* src, int K, int N) {
        int tot = K * N;
        k_transcvt<<<(tot + 255) / 256, 256, 0, stream>>>(dst, src, K, N);
    };
    auto cmb = [&](u16* dst, const float* wr, const float* wi, int K, int N) {
        int tot = 4 * K * N;
        k_combine<<<(tot + 255) / 256, 256, 0, stream>>>(dst, wr, wi, K, N);
    };

    // ---- phase 0: prep ----
    k_prep_ygi<<<1, 64, 0, stream>>>(YG, YGI);
    tc(ed_w1T, P(3), 768, 128);
    tc(ed_w2T, P(5), 128, 384);
    for (int l = 0; l < 2; l++) {
        tc(radw1T[l], LP(l, 4), 768, 128);
        tc(c1w0T[l],  LP(l, 8), 768, 640);
        cmb(c1c1T[l], LP(l, 9),  LP(l, 10), 512, 256);
        cmb(c1c2T[l], LP(l, 11), LP(l, 12), 256, 128);
        tc(c2w0T[l],  LP(l, 13), 384, 384);
        cmb(c2c1T[l], LP(l, 14), LP(l, 15), 256, 256);
        cmb(c2c2T[l], LP(l, 16), LP(l, 17), 128, 128);
        tc(aw1T[l], LP(l, 22), 128, 128);
        tc(aw2T[l], LP(l, 24), 128, 128);
    }

    // ---- phase 1: geometry + embeds + init x ----
    k_geom<<<(E + 255) / 256, 256, 0, stream>>>(ev, rot_gamma, Dbuf, envb, dbuf, E);
    k_embeds<<<E, 128, 0, stream>>>(dbuf, senders, receivers, species, P(1), P(2), EE, E);
    k_init_x<<<Nn, 128, 0, stream>>>(x, species, P(0), csd, n_node, NB, Nn);

    // ---- phase 2: initial message ----
    {
        off = arena0;
        u16*   hh  = (u16*)  alloc((size_t)E * 128 * 2);
        float* m0f = (float*)alloc((size_t)E * 384 * 4);
        gemm(EE, ed_w1T, P(4), nullptr, hh, E, 128, 768, 1);
        gemm(hh, ed_w2T, P(6), m0f, nullptr, E, 384, 128, 0);
        k_msg0<<<E, 128, 0, stream>>>(m0f, Dbuf, envb, receivers, x, E);
    }

    const int ECH = 10000;
    const int NCH = 2500;

    // ---- layers ----
    for (int l = 0; l < 2; l++) {
        // edgewise
        k_lnsh<<<Nn, 128, 0, stream>>>(x, xln, LP(l,0), LP(l,1), LP(l,2), LP(l,3), Nn);
        off = arena0;
        float* radhid = (float*)alloc((size_t)ECH * 128  * 4);
        float* rsb    = (float*)alloc((size_t)ECH * 9    * 4);
        u16*   A0     = (u16*)  alloc((size_t)ECH * 768  * 2);
        u16*   A1     = (u16*)  alloc((size_t)ECH * 1024 * 2);
        u16*   A2     = (u16*)  alloc((size_t)ECH * 512  * 2);
        float* y0f    = (float*)alloc((size_t)ECH * 640  * 4);
        float* p1o    = (float*)alloc((size_t)ECH * 512  * 4);
        float* p2o    = (float*)alloc((size_t)ECH * 256  * 4);
        u16*   hA0    = (u16*)  alloc((size_t)ECH * 384  * 2);
        u16*   hA1    = (u16*)  alloc((size_t)ECH * 512  * 2);
        u16*   hA2    = (u16*)  alloc((size_t)ECH * 256  * 2);
        float* c2y0   = (float*)alloc((size_t)ECH * 384  * 4);
        float* c2p1   = (float*)alloc((size_t)ECH * 512  * 4);
        float* c2p2   = (float*)alloc((size_t)ECH * 256  * 4);
        for (int e0 = 0; e0 < E; e0 += ECH) {
            int ecnt = (E - e0 < ECH) ? (E - e0) : ECH;
            gemm(EE + (size_t)e0 * 768, radw1T[l], LP(l,5), radhid, nullptr, ecnt, 128, 768, 1);
            k_rad2<<<(ecnt + 127) / 128, 128, 0, stream>>>(radhid, LP(l,6), LP(l,7), rsb, ecnt);
            k_edgewise<<<ecnt, 128, 0, stream>>>(xln, Dbuf, rsb, senders, receivers, e0, A0, A1, A2, ecnt);
            gemm(A0, c1w0T[l], nullptr, y0f, nullptr, ecnt, 640, 768, 0);
            gemm(A1, c1c1T[l], nullptr, p1o, nullptr, ecnt, 512, 1024, 0);
            gemm(A2, c1c2T[l], nullptr, p2o, nullptr, ecnt, 256, 512, 0);
            k_nonlin<<<ecnt, 128, 0, stream>>>(y0f, p1o, p2o, hA0, hA1, hA2, ecnt);
            gemm(hA0, c2w0T[l], nullptr, c2y0, nullptr, ecnt, 384, 384, 0);
            gemm(hA1, c2c1T[l], nullptr, c2p1, nullptr, ecnt, 512, 512, 0);
            gemm(hA2, c2c2T[l], nullptr, c2p2, nullptr, ecnt, 256, 256, 0);
            k_msg_scatter<<<ecnt, 128, 0, stream>>>(c2y0, c2p1, c2p2, Dbuf, envb, receivers, x, e0, ecnt);
        }
        // atomwise
        k_lnsh<<<Nn, 128, 0, stream>>>(x, xln, LP(l,18), LP(l,19), LP(l,20), LP(l,21), Nn);
        off = arena0;
        u16*   gb   = (u16*)  alloc((size_t)NCH * 64 * 128 * 2);
        u16*   midb = (u16*)  alloc((size_t)NCH * 64 * 128 * 2);
        float* gout = (float*)alloc((size_t)NCH * 64 * 128 * 4);
        for (int n0 = 0; n0 < Nn; n0 += NCH) {
            int ncnt = (Nn - n0 < NCH) ? (Nn - n0) : NCH;
            k_yg<<<ncnt, 128, 0, stream>>>(xln, n0, YG, gb, ncnt);
            gemm(gb,   aw1T[l], LP(l,23), nullptr, midb, ncnt * 64, 128, 128, 1);
            gemm(midb, aw2T[l], LP(l,25), gout, nullptr, ncnt * 64, 128, 128, 0);
            k_ygi<<<ncnt, 128, 0, stream>>>(gout, YGI, x, n0, ncnt);
        }
    }

    // ---- final layer norm (in place on d_out) ----
    k_lnsh<<<Nn, 128, 0, stream>>>(x, x, P(7), P(8), P(9), P(10), Nn);
}
